// GraphSAGE2_23699629539649
// MI455X (gfx1250) — matrix operand order repaired and checked by reading, not compiled, not run on hardware
//
#include <hip/hip_runtime.h>
#include <cstdint>

#define NNODES 10000
#define NEDGES 320000
#define HIDD   256
#define CH     64
#define LSEQ   140
#define INFEATS (CH*LSEQ)   // 8960
#define BNEPS  1e-5f

typedef _Float16 h16;
typedef h16   v16h __attribute__((ext_vector_type(16)));
typedef h16   v8h  __attribute__((ext_vector_type(8)));
typedef float v8f  __attribute__((ext_vector_type(8)));

// ---------------------------------------------------------------------------
// generic helpers
// ---------------------------------------------------------------------------
__global__ void zero_f32(float* __restrict__ p, int n) {
  int i = blockIdx.x * blockDim.x + threadIdx.x;
  if (i < n) p[i] = 0.f;
}

// W is (K x Ncols) fp32 row-major; Bt is (Ncols x K) f16 row-major (transposed,
// so each WMMA B-fragment lane load is 32B contiguous).
__global__ void pack_wT_f16(const float* __restrict__ W, h16* __restrict__ Bt,
                            int K, int Ncols) {
  int i = blockIdx.x * blockDim.x + threadIdx.x;
  if (i >= K * Ncols) return;
  int n = i / K, k = i - n * K;
  Bt[(size_t)n * K + k] = (h16)W[(size_t)k * Ncols + n];
}

// ---------------------------------------------------------------------------
// f16 WMMA GEMM: C(MxNcols,f32) = A(MxK,f16 row-major) * Bt^T (Bt: Ncols x K)
// wave -> 16 rows x 64 cols; block = 8 waves stacked in M; K % 32 == 0.
// ---------------------------------------------------------------------------
__global__ __launch_bounds__(256)
void gemm_f16_wmma(const h16* __restrict__ A, const h16* __restrict__ Bt,
                   float* __restrict__ C, int M, int K, int Ncols) {
  const int lane = threadIdx.x & 31;
  const int wave = threadIdx.x >> 5;
  const int row0 = (blockIdx.x * 8 + wave) * 16;
  const int col0 = blockIdx.y * 64;
  if (row0 >= M) return;                 // whole-wave uniform exit (EXEC all-1 inside)

  const int half = lane >> 4;            // 0 / 1
  const int m    = lane & 15;
  // A-frag: lane holds row m, K = half*16 .. half*16+15 (contiguous, matches B)
  const h16* pa = A + (size_t)(row0 + m) * K + half * 16;
  // B-frag: lane holds col n=lane&15, K = half*16 .. half*16+15 (contiguous in Bt)
  const h16* pb = Bt + (size_t)(col0 + m) * K + half * 16;

  v8f acc0 = {}, acc1 = {}, acc2 = {}, acc3 = {};
  for (int k0 = 0; k0 < K; k0 += 32) {
    __builtin_prefetch(pa + k0 + 128, 0, 0);     // stream A ahead (global_prefetch_b8)
    v8h alo = *(const v8h*)(pa + k0);
    v8h ahi = *(const v8h*)(pa + k0 + 8);
    v16h a  = __builtin_shufflevector(alo, ahi, 0,1,2,3,4,5,6,7,8,9,10,11,12,13,14,15);
    v16h b0 = *(const v16h*)(pb + k0);
    v16h b1 = *(const v16h*)(pb + (size_t)16 * K + k0);
    v16h b2 = *(const v16h*)(pb + (size_t)32 * K + k0);
    v16h b3 = *(const v16h*)(pb + (size_t)48 * K + k0);
    acc0 = __builtin_amdgcn_wmma_f32_16x16x32_f16(false, a, false, b0, (short)0, acc0, false, false);
    acc1 = __builtin_amdgcn_wmma_f32_16x16x32_f16(false, a, false, b1, (short)0, acc1, false, false);
    acc2 = __builtin_amdgcn_wmma_f32_16x16x32_f16(false, a, false, b2, (short)0, acc2, false, false);
    acc3 = __builtin_amdgcn_wmma_f32_16x16x32_f16(false, a, false, b3, (short)0, acc3, false, false);
  }
  // C/D layout: lane L -> col L&15, rows half*8 + r
  float* cp = C + (size_t)(row0 + half * 8) * Ncols + col0 + m;
#pragma unroll
  for (int r = 0; r < 8; ++r) {
    cp[(size_t)r * Ncols +  0] = acc0[r];
    cp[(size_t)r * Ncols + 16] = acc1[r];
    cp[(size_t)r * Ncols + 32] = acc2[r];
    cp[(size_t)r * Ncols + 48] = acc3[r];
  }
}

// ---------------------------------------------------------------------------
// conv1: out[n,o,l] = sum_c x[n,c,l]*w1[o,c] + b1[o]   (c=5)
// two-pass BN-over-channels (recompute instead of materializing pre-BN tensor)
// ---------------------------------------------------------------------------
__global__ __launch_bounds__(256)
void conv1_stats(const float* __restrict__ x, const float* __restrict__ w,
                 const float* __restrict__ b, float* __restrict__ stat) {
  const int o = threadIdx.x & 63;
  const int sub = threadIdx.x >> 6;
  const float w0 = w[o*5+0], w1 = w[o*5+1], w2 = w[o*5+2], w3 = w[o*5+3], w4 = w[o*5+4];
  const float bo = b[o];
  float s = 0.f, s2 = 0.f;
  const int P = NNODES * LSEQ;
  for (int p = blockIdx.x * 4 + sub; p < P; p += gridDim.x * 4) {
    int n = p / LSEQ, l = p - n * LSEQ;
    const float* xp = x + (size_t)n * 5 * LSEQ + l;
    float v = bo + w0*xp[0] + w1*xp[LSEQ] + w2*xp[2*LSEQ] + w3*xp[3*LSEQ] + w4*xp[4*LSEQ];
    s += v; s2 += v * v;
  }
  __shared__ float sh[256], sh2[256];
  sh[threadIdx.x] = s; sh2[threadIdx.x] = s2;
  __syncthreads();
  if (sub == 0) {
    float a = sh[o] + sh[o+64] + sh[o+128] + sh[o+192];
    float c = sh2[o] + sh2[o+64] + sh2[o+128] + sh2[o+192];
    atomicAdd(&stat[o], a);
    atomicAdd(&stat[64 + o], c);
  }
}

__global__ void conv1_apply(const float* __restrict__ x, const float* __restrict__ w,
                            const float* __restrict__ b, const float* __restrict__ scale,
                            const float* __restrict__ shift, float* __restrict__ cnn1) {
  int idx = blockIdx.x * blockDim.x + threadIdx.x;
  if (idx >= NNODES * INFEATS) return;
  int n = idx / INFEATS;
  int rem = idx - n * INFEATS;
  int o = rem / LSEQ;
  int l = rem - o * LSEQ;
  const float* xp = x + (size_t)n * 5 * LSEQ + l;
  float v = b[o] + w[o*5+0]*xp[0] + w[o*5+1]*xp[LSEQ] + w[o*5+2]*xp[2*LSEQ]
                 + w[o*5+3]*xp[3*LSEQ] + w[o*5+4]*xp[4*LSEQ];
  cnn1[idx] = fmaxf(v * scale[o] + shift[o], 0.f);
}

// ---------------------------------------------------------------------------
// conv2: out[n,o,l] = sum_{c<64} h1[n,c,l]*w2[o,c] + b2[o]
// ---------------------------------------------------------------------------
__global__ __launch_bounds__(256)
void conv2_stats(const float* __restrict__ h1, const float* __restrict__ w2,
                 const float* __restrict__ b2, float* __restrict__ stat) {
  const int o = threadIdx.x & 63;
  const int sub = threadIdx.x >> 6;
  const float bo = b2[o];
  float s = 0.f, s2 = 0.f;
  const int P = NNODES * LSEQ;
  for (int p = blockIdx.x * 4 + sub; p < P; p += gridDim.x * 4) {
    int n = p / LSEQ, l = p - n * LSEQ;
    const float* hp = h1 + (size_t)n * INFEATS + l;
    float v = bo;
#pragma unroll 8
    for (int c = 0; c < CH; ++c) v += w2[o*CH + c] * hp[(size_t)c * LSEQ];
    s += v; s2 += v * v;
  }
  __shared__ float sh[256], sh2[256];
  sh[threadIdx.x] = s; sh2[threadIdx.x] = s2;
  __syncthreads();
  if (sub == 0) {
    float a = sh[o] + sh[o+64] + sh[o+128] + sh[o+192];
    float c = sh2[o] + sh2[o+64] + sh2[o+128] + sh2[o+192];
    atomicAdd(&stat[o], a);
    atomicAdd(&stat[64 + o], c);
  }
}

__global__ void conv2_apply(const float* __restrict__ h1, const float* __restrict__ w2,
                            const float* __restrict__ b2, const float* __restrict__ scale,
                            const float* __restrict__ shift, float* __restrict__ cnn2,
                            h16* __restrict__ hf16) {
  int idx = blockIdx.x * blockDim.x + threadIdx.x;
  if (idx >= NNODES * INFEATS) return;
  int n = idx / INFEATS;
  int rem = idx - n * INFEATS;
  int o = rem / LSEQ;
  int l = rem - o * LSEQ;
  const float* hp = h1 + (size_t)n * INFEATS + l;
  float v = b2[o];
#pragma unroll 8
  for (int c = 0; c < CH; ++c) v += w2[o*CH + c] * hp[(size_t)c * LSEQ];
  float r = fmaxf(v * scale[o] + shift[o], 0.f) + h1[idx];   // residual
  cnn2[idx] = r;
  hf16[idx] = (h16)r;                                        // reshape(N,-1) is identity here
}

// ---------------------------------------------------------------------------
// graph degree + aggregation
// ---------------------------------------------------------------------------
__global__ void degree_kernel(const int* __restrict__ edges, float* __restrict__ deg, int E) {
  int e = blockIdx.x * blockDim.x + threadIdx.x;
  if (e < E) atomicAdd(&deg[edges[2*e + 1]], 1.0f);
}

__global__ void invdenom_kernel(float* __restrict__ deg, int n) {
  int i = blockIdx.x * blockDim.x + threadIdx.x;
  if (i < n) deg[i] = 1.0f / fmaxf(deg[i], 1.0f);
}

// hshn: N x 512 (cols 0..255 = h@ws, cols 256..511 = h@wn). Adds hn[src]/deg[dst]
// into the hs half. Read-cols and write-cols are disjoint -> safe.
__global__ __launch_bounds__(256)
void sage_aggregate(const int* __restrict__ edges, const float* __restrict__ inv,
                    float* __restrict__ hshn) {
  int e = blockIdx.x;
  int src = edges[2*e];
  int dst = edges[2*e + 1];
  float v = hshn[(size_t)src * 512 + 256 + threadIdx.x] * inv[dst];
  atomicAdd(&hshn[(size_t)dst * 512 + threadIdx.x], v);
}

// ---------------------------------------------------------------------------
// BN-over-nodes (per column): one block per column, value = raw + bias[col]
// ---------------------------------------------------------------------------
__global__ __launch_bounds__(256)
void bn_node_stats(const float* __restrict__ X, int ldx, int rows,
                   const float* __restrict__ bias, float* __restrict__ stat, int C) {
  int col = blockIdx.x;
  float bcol = bias[col];
  float s = 0.f, s2 = 0.f;
  for (int r = threadIdx.x; r < rows; r += 256) {
    float v = X[(size_t)r * ldx + col] + bcol;
    s += v; s2 += v * v;
  }
  __shared__ float sh[256], sh2[256];
  sh[threadIdx.x] = s; sh2[threadIdx.x] = s2;
  __syncthreads();
  for (int off = 128; off; off >>= 1) {
    if (threadIdx.x < off) { sh[threadIdx.x] += sh[threadIdx.x+off]; sh2[threadIdx.x] += sh2[threadIdx.x+off]; }
    __syncthreads();
  }
  if (threadIdx.x == 0) { stat[col] = sh[0]; stat[C + col] = sh2[0]; }
}

__global__ void bn_finalize(const float* __restrict__ stat, int C, float invCnt,
                            const float* __restrict__ g, const float* __restrict__ bb,
                            float* __restrict__ scale, float* __restrict__ shift) {
  int i = blockIdx.x * blockDim.x + threadIdx.x;
  if (i >= C) return;
  float m = stat[i] * invCnt;
  float v = stat[C + i] * invCnt - m * m;
  float sc = g[i] * rsqrtf(v + BNEPS);
  scale[i] = sc;
  shift[i] = bb[i] - m * sc;
}

__global__ void sage_apply(const float* __restrict__ hshn, const float* __restrict__ bias,
                           const float* __restrict__ scale, const float* __restrict__ shift,
                           const float* __restrict__ hprev, float* __restrict__ hnew,
                           h16* __restrict__ h16out, float* __restrict__ finout, int n) {
  int idx = blockIdx.x * blockDim.x + threadIdx.x;
  if (idx >= n * HIDD) return;
  int row = idx >> 8, col = idx & 255;
  float v = (hshn[(size_t)row * 512 + col] + bias[col]) * scale[col] + shift[col];
  if (hprev) v += hprev[idx];
  v = fmaxf(v, 0.f);
  hnew[idx] = v;
  h16out[idx] = (h16)v;
  if (finout) finout[idx] = v;
}

__global__ void mlp_final(const float* __restrict__ m1, const float* __restrict__ b1,
                          const float* __restrict__ scale, const float* __restrict__ shift,
                          const float* __restrict__ w2, const float* __restrict__ b2,
                          float* __restrict__ out, int n) {
  int i = blockIdx.x * blockDim.x + threadIdx.x;
  if (i >= n) return;
  const float* r = m1 + (size_t)i * CH;
  float o0 = b2[0], o1 = b2[1];
#pragma unroll 8
  for (int o = 0; o < CH; ++o) {
    float z = fmaxf((r[o] + b1[o]) * scale[o] + shift[o], 0.f);
    o0 += z * w2[o*2 + 0];
    o1 += z * w2[o*2 + 1];
  }
  out[i*2 + 0] = o0;
  out[i*2 + 1] = o1;
}

// ---------------------------------------------------------------------------
// orchestration
// ---------------------------------------------------------------------------
extern "C" void kernel_launch(void* const* d_in, const int* in_sizes, int n_in,
                              void* d_out, int out_size, void* d_ws, size_t ws_size,
                              hipStream_t stream) {
  (void)in_sizes; (void)n_in; (void)out_size; (void)ws_size;
  const float* x       = (const float*)d_in[0];
  const float* conv1w  = (const float*)d_in[1];
  const float* conv1b  = (const float*)d_in[2];
  const float* bn_c1_g = (const float*)d_in[3];
  const float* bn_c1_b = (const float*)d_in[4];
  const float* conv2w  = (const float*)d_in[5];
  const float* conv2b  = (const float*)d_in[6];
  const float* bn_c2_g = (const float*)d_in[7];
  const float* bn_c2_b = (const float*)d_in[8];
  const float* ws0     = (const float*)d_in[9];
  const float* wn0     = (const float*)d_in[10];
  const float* b0      = (const float*)d_in[11];
  const float* ws_rest = (const float*)d_in[12];
  const float* wn_rest = (const float*)d_in[13];
  const float* b_rest  = (const float*)d_in[14];
  const float* bn_g    = (const float*)d_in[15];
  const float* bn_b    = (const float*)d_in[16];
  const float* mlp_w1  = (const float*)d_in[17];
  const float* mlp_b1  = (const float*)d_in[18];
  const float* mlp_bng = (const float*)d_in[19];
  const float* mlp_bnb = (const float*)d_in[20];
  const float* mlp_w2  = (const float*)d_in[21];
  const float* mlp_b2  = (const float*)d_in[22];
  const int*   edges   = (const int*)d_in[23];

  // output regions (flat concat: out, cnn1, cnn2, fin)
  float* outp = (float*)d_out;
  float* cnn1 = outp + (size_t)NNODES * 2;
  float* cnn2 = cnn1 + (size_t)NNODES * INFEATS;
  float* fin  = cnn2 + (size_t)NNODES * INFEATS;

  // workspace carve (256B aligned)
  uintptr_t base = (uintptr_t)d_ws;
  auto carve = [&](size_t bytes) -> void* {
    uintptr_t p = (base + 255) & ~(uintptr_t)255;
    base = p + bytes;
    return (void*)p;
  };
  h16*   P0    = (h16*)carve((size_t)512 * INFEATS * 2);   // packed [ws0;wn0]^T
  h16*   PR    = (h16*)carve((size_t)3 * 512 * HIDD * 2);  // packed layer 1-3 weights
  h16*   PM    = (h16*)carve((size_t)CH * HIDD * 2);       // packed mlp_w1^T
  h16*   HCONV = (h16*)carve((size_t)NNODES * INFEATS * 2);// f16 copy of cnn2 (layer0 A)
  float* HSHN  = (float*)carve((size_t)NNODES * 512 * 4);  // [h@ws | h@wn]
  float* HF32A = (float*)carve((size_t)NNODES * HIDD * 4);
  float* HF32B = (float*)carve((size_t)NNODES * HIDD * 4);
  h16*   HS16  = (h16*)carve((size_t)NNODES * HIDD * 2);   // f16 h for next GEMM
  float* M1    = (float*)carve((size_t)NNODES * CH * 4);
  float* DEG   = (float*)carve((size_t)NNODES * 4);
  float* STAT  = (float*)carve(512 * 4);
  float* SCALE = (float*)carve(256 * 4);
  float* SHIFT = (float*)carve(256 * 4);

  // ---- pack weights (transposed f16) ----
  {
    int t = INFEATS * HIDD;
    pack_wT_f16<<<(t + 255) / 256, 256, 0, stream>>>(ws0, P0, INFEATS, HIDD);
    pack_wT_f16<<<(t + 255) / 256, 256, 0, stream>>>(wn0, P0 + (size_t)HIDD * INFEATS, INFEATS, HIDD);
    t = HIDD * HIDD;
    for (int l = 0; l < 3; ++l) {
      pack_wT_f16<<<(t + 255) / 256, 256, 0, stream>>>(
          ws_rest + (size_t)l * HIDD * HIDD, PR + (size_t)l * 512 * HIDD, HIDD, HIDD);
      pack_wT_f16<<<(t + 255) / 256, 256, 0, stream>>>(
          wn_rest + (size_t)l * HIDD * HIDD, PR + (size_t)l * 512 * HIDD + (size_t)HIDD * HIDD, HIDD, HIDD);
    }
    t = HIDD * CH;
    pack_wT_f16<<<(t + 255) / 256, 256, 0, stream>>>(mlp_w1, PM, HIDD, CH);
  }

  const int elemsConv = NNODES * INFEATS;

  // ---- conv1 + bn + relu -> cnn1 ----
  zero_f32<<<1, 256, 0, stream>>>(STAT, 128);
  conv1_stats<<<1024, 256, 0, stream>>>(x, conv1w, conv1b, STAT);
  bn_finalize<<<1, 64, 0, stream>>>(STAT, 64, 1.0f / (NNODES * LSEQ), bn_c1_g, bn_c1_b, SCALE, SHIFT);
  conv1_apply<<<(elemsConv + 255) / 256, 256, 0, stream>>>(x, conv1w, conv1b, SCALE, SHIFT, cnn1);

  // ---- conv2 + bn + relu + residual -> cnn2 (+ f16 copy) ----
  zero_f32<<<1, 256, 0, stream>>>(STAT, 128);
  conv2_stats<<<2048, 256, 0, stream>>>(cnn1, conv2w, conv2b, STAT);
  bn_finalize<<<1, 64, 0, stream>>>(STAT, 64, 1.0f / (NNODES * LSEQ), bn_c2_g, bn_c2_b, SCALE, SHIFT);
  conv2_apply<<<(elemsConv + 255) / 256, 256, 0, stream>>>(cnn1, conv2w, conv2b, SCALE, SHIFT, cnn2, HCONV);

  // ---- degree / inverse denom ----
  zero_f32<<<(NNODES + 255) / 256, 256, 0, stream>>>(DEG, NNODES);
  degree_kernel<<<(NEDGES + 255) / 256, 256, 0, stream>>>(edges, DEG, NEDGES);
  invdenom_kernel<<<(NNODES + 255) / 256, 256, 0, stream>>>(DEG, NNODES);

  // ---- 4 SAGE layers ----
  dim3 gGemm((625 + 7) / 8, 8);   // 10000/16 = 625 row tiles; 512 cols / 64
  float* hprev = nullptr;
  float* hcur  = HF32A;
  for (int l = 0; l < 4; ++l) {
    const h16* Amat = (l == 0) ? HCONV : HS16;
    int K = (l == 0) ? INFEATS : HIDD;
    const h16* Bt = (l == 0) ? P0 : (PR + (size_t)(l - 1) * 512 * HIDD);
    gemm_f16_wmma<<<gGemm, 256, 0, stream>>>(Amat, Bt, HSHN, NNODES, K, 512);
    sage_aggregate<<<NEDGES, 256, 0, stream>>>(edges, DEG, HSHN);
    const float* bias = (l == 0) ? b0 : (b_rest + (size_t)(l - 1) * HIDD);
    bn_node_stats<<<HIDD, 256, 0, stream>>>(HSHN, 512, NNODES, bias, STAT, HIDD);
    bn_finalize<<<1, 256, 0, stream>>>(STAT, HIDD, 1.0f / NNODES,
                                       bn_g + (size_t)l * HIDD, bn_b + (size_t)l * HIDD, SCALE, SHIFT);
    sage_apply<<<(NNODES * HIDD + 255) / 256, 256, 0, stream>>>(
        HSHN, bias, SCALE, SHIFT, hprev, hcur, HS16, (l == 3) ? fin : nullptr, NNODES);
    hprev = hcur;
    hcur = (hcur == HF32A) ? HF32B : HF32A;
  }

  // ---- MLP head ----
  gemm_f16_wmma<<<dim3((625 + 7) / 8, 1), 256, 0, stream>>>(HS16, PM, M1, NNODES, HIDD, CH);
  bn_node_stats<<<CH, 256, 0, stream>>>(M1, CH, NNODES, mlp_b1, STAT, CH);
  bn_finalize<<<1, 64, 0, stream>>>(STAT, CH, 1.0f / NNODES, mlp_bng, mlp_bnb, SCALE, SHIFT);
  mlp_final<<<(NNODES + 255) / 256, 256, 0, stream>>>(M1, mlp_b1, SCALE, SHIFT, mlp_w2, mlp_b2, outp, NNODES);
}
